// VariationalRNNEncoder_50766513439373
// MI455X (gfx1250) — compile-verified, matrix operand order checked by reading, not run on hardware
//
#include <hip/hip_runtime.h>
#include <hip/hip_bf16.h>

// Sizes from the reference
#define Bb 512
#define Tt 512
#define Dd 64
#define Hh 256
#define Zz 64

// Per-workgroup batch rows. BM=64 => 8 WGs; each weight byte fetched from L2
// feeds 64 batch rows per step (4x less total L2 traffic than BM=16).
#define BM 64

typedef __attribute__((ext_vector_type(16))) _Float16 v16h;
typedef __attribute__((ext_vector_type(8)))  _Float16 v8h;
typedef __attribute__((ext_vector_type(8)))  float    v8f;

// ---------------------------------------------------------------------------
// WMMA fragment load: works for BOTH A (activations, row-major [16 x K]) and
// B (weights, row-major [16 rows of N x K]) operands of D = A * W^T.
// Per CDNA5 ISA 16-bit A layout: lane L holds row (L%16); VGPRs 0-3 hold
// K = kb..kb+7, VGPRs 4-7 hold K = kb+16..kb+23, where kb = k0 + (L>=16 ? 8:0).
// ---------------------------------------------------------------------------
__device__ inline v16h frag_ld(const _Float16* tile, int ldk, int k0, int lane) {
  const int m  = lane & 15;
  const int kb = k0 + ((lane & 16) ? 8 : 0);
  const _Float16* p = tile + m * ldk + kb;
  v8h lo = *reinterpret_cast<const v8h*>(p);
  v8h hi = *reinterpret_cast<const v8h*>(p + 16);
  v16h f;
#pragma unroll
  for (int i = 0; i < 8; ++i) { f[i] = lo[i]; f[i + 8] = hi[i]; }
  return f;
}

// Accumulating 16x16 tile GEMM: acc += A(16xK) * W(16 rows x K)^T
template <int K>
__device__ inline v8f gemm_acc(v8f acc, const _Float16* A, int lda,
                               const _Float16* W, int ldw, int lane) {
#pragma unroll
  for (int k = 0; k < K; k += 32) {
    v16h a = frag_ld(A, lda, k, lane);
    v16h b = frag_ld(W, ldw, k, lane);
    acc = __builtin_amdgcn_wmma_f32_16x16x32_f16(
        /*neg_a=*/false, a, /*neg_b=*/false, b,
        /*c_mod=*/(short)0, acc, /*reuse_a=*/false, /*reuse_b=*/false);
  }
  return acc;
}

// Epilogue: bias + ReLU, store f16 tile to LDS. C/D layout: lane L holds
// column n0+(L&15); VGPR r holds row r + ((L&16)?8:0).
__device__ inline void store_relu_h(_Float16* dst, int ldd, int n0, v8f acc,
                                    const float* bias, int lane) {
  const int n  = n0 + (lane & 15);
  const int mb = (lane & 16) ? 8 : 0;
  const float bv = bias[n];
#pragma unroll
  for (int r = 0; r < 8; ++r)
    dst[(mb + r) * ldd + n] = (_Float16)fmaxf(acc[r] + bv, 0.0f);
}

__device__ inline float sigmoidf_(float v) { return 1.0f / (1.0f + expf(-v)); }
__device__ inline float softplusf_(float v) {
  return log1pf(expf(-fabsf(v))) + fmaxf(v, 0.0f);
}

// ---------------------------------------------------------------------------
// Weight f32 -> f16 conversion (weights stay hot in 192MB L2 afterwards)
// ---------------------------------------------------------------------------
__global__ void f32_to_f16_kernel(const float* __restrict__ s,
                                  _Float16* __restrict__ d, int n) {
  int i = (blockIdx.x * blockDim.x + threadIdx.x) * 4;
  if (i + 3 < n) {
    float4 v = *reinterpret_cast<const float4*>(s + i);
    d[i + 0] = (_Float16)v.x; d[i + 1] = (_Float16)v.y;
    d[i + 2] = (_Float16)v.z; d[i + 3] = (_Float16)v.w;
  }
}

// ---------------------------------------------------------------------------
// Persistent VRNN recurrence. 8 WGs x 512 threads (16 waves). Each WG owns 64
// batch rows for all T=512 steps. Wave = (column-group cg 0..7, M-half mh):
// it computes columns [cg*32, cg*32+32) for M-tiles {2mh, 2mh+1}, M-tile loop
// rolled so only ~4 v8f accumulators are ever live (no spills). The f32
// master copy of h lives in per-wave registers (hreg[2][2]); the f16 GEMM
// copy of h is double-buffered in LDS. A single integer offset is laundered
// through an empty asm each step: weight addresses become loop-variant (so
// LICM cannot hoist+spill the weight loads) while pointer provenance is kept
// (loads stay global_load, tracked by LOADcnt only — not FLAT/DScnt).
// 192KB dynamic LDS (CDNA5 WGP supports up to 320KB per workgroup).
// ---------------------------------------------------------------------------
__global__ __launch_bounds__(512)
void vrnn_kernel(const float* __restrict__ x, const float* __restrict__ h0,
                 const float* __restrict__ eps,
                 const float* __restrict__ b_x1, const float* __restrict__ b_x2,
                 const float* __restrict__ b_z,  const float* __restrict__ b_e1,
                 const float* __restrict__ b_e2, const float* __restrict__ b_em,
                 const float* __restrict__ b_es,
                 const _Float16* __restrict__ wx1_, const _Float16* __restrict__ wx2_,
                 const _Float16* __restrict__ wz_,  const _Float16* __restrict__ we1_,
                 const _Float16* __restrict__ we2_, const _Float16* __restrict__ wem_,
                 const _Float16* __restrict__ wes_, const _Float16* __restrict__ wih_,
                 const _Float16* __restrict__ whh_,
                 float* __restrict__ out) {
  extern __shared__ __align__(16) unsigned char smem[];
  _Float16* phix = (_Float16*)(smem);              // [64][256] persistent
  _Float16* h16a = (_Float16*)(smem + 32768);      // [64][256] h f16, buffer A
  _Float16* h16b = (_Float16*)(smem + 65536);      // [64][256] h f16, buffer B
  _Float16* phiz = (_Float16*)(smem + 98304);      // [64][256] persistent
  unsigned char* tr = smem + 131072;               // 64KB transient, aliased
  _Float16* xbuf  = (_Float16*)(tr);               // [64][64]   phase A
  _Float16* a1    = (_Float16*)(tr + 8192);        // [64][256]  phase A
  _Float16* enc1  = (_Float16*)(tr);               // [64][256]  phase B
  _Float16* enc2  = (_Float16*)(tr + 32768);       // [64][256]  phase B/C
  float*    meanb = (float*)(tr);                  // [64][64]   phase C
  float*    stdb  = (float*)(tr + 16384);          // [64][64]   phase C
  _Float16* zbuf  = (_Float16*)(tr + 32768);       // [64][64]   phase C/D (over dead enc2)

  const int tid  = threadIdx.x;
  const int wv   = tid >> 5;             // 0..15
  const int lane = tid & 31;
  const int cg   = wv & 7;               // column group: columns [cg*32, cg*32+32)
  const int mh   = wv >> 3;              // M-half: M-tiles {2mh, 2mh+1}
  const int r0   = blockIdx.x * BM;
  const int cn   = lane & 15;            // column-within-tile (C/D layout)
  const int mb   = (lane & 16) ? 8 : 0;  // row base (C/D layout)

  // f32 master copy of h in registers:
  // hreg[ml][i][r] <-> h[m = (2mh+ml)*16 + mb + r][n = cg*32 + i*16 + cn]
  v8f hreg[2][2];
#pragma unroll
  for (int ml = 0; ml < 2; ++ml)
#pragma unroll
    for (int i = 0; i < 2; ++i) {
      int n = cg * 32 + i * 16 + cn;
      int m0 = (mh * 2 + ml) * 16 + mb;
#pragma unroll
      for (int r = 0; r < 8; ++r)
        hreg[ml][i][r] = h0[(size_t)(r0 + m0 + r) * Hh + n];
    }
  // init f16 h buffer A
  for (int e = tid; e < BM * 256; e += 512) {
    int m = e >> 8, c = e & 255;
    h16a[e] = (_Float16)h0[(size_t)(r0 + m) * Hh + c];
  }
  __syncthreads();

  long woff = 0;  // laundered each step: blocks LICM, keeps global provenance
#pragma unroll 1
  for (int t = 0; t < Tt; ++t) {
    asm volatile("" : "+s"(woff));
    const _Float16* wx1 = wx1_ + woff;
    const _Float16* wx2 = wx2_ + woff;
    const _Float16* wz  = wz_  + woff;
    const _Float16* we1 = we1_ + woff;
    const _Float16* we2 = we2_ + woff;
    const _Float16* wem = wem_ + woff;
    const _Float16* wes = wes_ + woff;
    const _Float16* wih = wih_ + woff;
    const _Float16* whh = whh_ + woff;

    const _Float16* hcur = (t & 1) ? h16b : h16a;
    _Float16*       hnxt = (t & 1) ? h16a : h16b;

    // ---- phase A: load x_t tile (+ prefetch t+1 x/eps), then
    //      phi_x = relu(relu(x Wx1^T+b) Wx2^T+b) ----
    for (int e = tid * 4; e < BM * 64; e += 512 * 4) {
      int m = e >> 6, c = e & 63;
      float4 xv = *reinterpret_cast<const float4*>(
          &x[((size_t)(r0 + m) * Tt + t) * Dd + c]);
      xbuf[e + 0] = (_Float16)xv.x; xbuf[e + 1] = (_Float16)xv.y;
      xbuf[e + 2] = (_Float16)xv.z; xbuf[e + 3] = (_Float16)xv.w;
      if (t + 1 < Tt) {
        __builtin_prefetch(&x[((size_t)(r0 + m) * Tt + (t + 1)) * Dd + c], 0, 1);
        __builtin_prefetch(&eps[(size_t)(t + 1) * Bb * Zz + (size_t)(r0 + m) * Zz + c], 0, 1);
      }
    }
    __syncthreads();
#pragma unroll 1
    for (int ml = 0; ml < 2; ++ml) {
      int mt = mh * 2 + ml;
#pragma unroll
      for (int i = 0; i < 2; ++i) {
        int n0 = cg * 32 + i * 16;
        v8f acc = {0, 0, 0, 0, 0, 0, 0, 0};
        acc = gemm_acc<64>(acc, xbuf + mt * 16 * 64, 64, wx1 + n0 * 64, 64, lane);
        store_relu_h(a1 + mt * 16 * 256, 256, n0, acc, b_x1, lane);
      }
    }
    __syncthreads();
#pragma unroll 1
    for (int ml = 0; ml < 2; ++ml) {
      int mt = mh * 2 + ml;
#pragma unroll
      for (int i = 0; i < 2; ++i) {
        int n0 = cg * 32 + i * 16;
        v8f acc = {0, 0, 0, 0, 0, 0, 0, 0};
        acc = gemm_acc<256>(acc, a1 + mt * 16 * 256, 256, wx2 + n0 * 256, 256, lane);
        store_relu_h(phix + mt * 16 * 256, 256, n0, acc, b_x2, lane);
      }
    }
    __syncthreads();

    // ---- phase B: enc1 = relu([phix|h] We1^T + b), enc2 = relu(enc1 We2^T + b)
#pragma unroll 1
    for (int ml = 0; ml < 2; ++ml) {
      int mt = mh * 2 + ml;
#pragma unroll
      for (int i = 0; i < 2; ++i) {
        int n0 = cg * 32 + i * 16;
        v8f acc = {0, 0, 0, 0, 0, 0, 0, 0};
        acc = gemm_acc<256>(acc, phix + mt * 16 * 256, 256, we1 + n0 * 512, 512, lane);
        acc = gemm_acc<256>(acc, hcur + mt * 16 * 256, 256, we1 + n0 * 512 + 256, 512, lane);
        store_relu_h(enc1 + mt * 16 * 256, 256, n0, acc, b_e1, lane);
      }
    }
    __syncthreads();
#pragma unroll 1
    for (int ml = 0; ml < 2; ++ml) {
      int mt = mh * 2 + ml;
#pragma unroll
      for (int i = 0; i < 2; ++i) {
        int n0 = cg * 32 + i * 16;
        v8f acc = {0, 0, 0, 0, 0, 0, 0, 0};
        acc = gemm_acc<256>(acc, enc1 + mt * 16 * 256, 256, we2 + n0 * 256, 256, lane);
        store_relu_h(enc2 + mt * 16 * 256, 256, n0, acc, b_e2, lane);
      }
    }
    __syncthreads();

    // ---- phase C: mean / softplus(std) (32 tiles = 2 per wave), then
    //      z = eps*std + mean ----
#pragma unroll 1
    for (int ml = 0; ml < 2; ++ml) {
      int mt = mh * 2 + ml;
      const _Float16* Ae = enc2 + mt * 16 * 256;
      if (cg < 4) {
        int n0 = cg * 16;
        v8f acc = {0, 0, 0, 0, 0, 0, 0, 0};
        acc = gemm_acc<256>(acc, Ae, 256, wem + n0 * 256, 256, lane);
        float bv = b_em[n0 + cn];
#pragma unroll
        for (int r = 0; r < 8; ++r)
          meanb[(mt * 16 + mb + r) * 64 + n0 + cn] = acc[r] + bv;
      } else {
        int n0 = (cg - 4) * 16;
        v8f acc = {0, 0, 0, 0, 0, 0, 0, 0};
        acc = gemm_acc<256>(acc, Ae, 256, wes + n0 * 256, 256, lane);
        float bv = b_es[n0 + cn];
#pragma unroll
        for (int r = 0; r < 8; ++r)
          stdb[(mt * 16 + mb + r) * 64 + n0 + cn] = softplusf_(acc[r] + bv);
      }
    }
    __syncthreads();
    for (int e = tid * 4; e < BM * 64; e += 512 * 4) {
      int m = e >> 6, c = e & 63;
      float4 ev = *reinterpret_cast<const float4*>(
          &eps[(size_t)t * Bb * Zz + (size_t)(r0 + m) * Zz + c]);
      zbuf[e + 0] = (_Float16)(ev.x * stdb[e + 0] + meanb[e + 0]);
      zbuf[e + 1] = (_Float16)(ev.y * stdb[e + 1] + meanb[e + 1]);
      zbuf[e + 2] = (_Float16)(ev.z * stdb[e + 2] + meanb[e + 2]);
      zbuf[e + 3] = (_Float16)(ev.w * stdb[e + 3] + meanb[e + 3]);
    }
    __syncthreads();

    // ---- phase D: phi_z = relu(z Wz^T + b_z) ----
#pragma unroll 1
    for (int ml = 0; ml < 2; ++ml) {
      int mt = mh * 2 + ml;
#pragma unroll
      for (int i = 0; i < 2; ++i) {
        int n0 = cg * 32 + i * 16;
        v8f acc = {0, 0, 0, 0, 0, 0, 0, 0};
        acc = gemm_acc<64>(acc, zbuf + mt * 16 * 64, 64, wz + n0 * 64, 64, lane);
        store_relu_h(phiz + mt * 16 * 256, 256, n0, acc, b_z, lane);
      }
    }
    __syncthreads();

    // ---- phase E: GRU gates. Wave owns the same 32 columns in all three
    // gate groups, so r/z/n for an (m,n) share a lane/VGPR slot and the
    // update runs in registers. h_new goes to hnxt, so no mid-phase barrier.
#pragma unroll 1
    for (int ml = 0; ml < 2; ++ml) {
      int mt = mh * 2 + ml;
      const _Float16* Ax = phix + mt * 16 * 256;
      const _Float16* Az = phiz + mt * 16 * 256;
      const _Float16* Ah = hcur + mt * 16 * 256;
#pragma unroll 1
      for (int i = 0; i < 2; ++i) {
        int n0 = cg * 32 + i * 16;
        v8f Sr = {0, 0, 0, 0, 0, 0, 0, 0};          // r: gi_r + gh_r
        Sr = gemm_acc<256>(Sr, Ax, 256, wih + n0 * 512, 512, lane);
        Sr = gemm_acc<256>(Sr, Az, 256, wih + n0 * 512 + 256, 512, lane);
        Sr = gemm_acc<256>(Sr, Ah, 256, whh + n0 * 256, 256, lane);
        v8f Sz = {0, 0, 0, 0, 0, 0, 0, 0};          // z: gi_z + gh_z
        Sz = gemm_acc<256>(Sz, Ax, 256, wih + (256 + n0) * 512, 512, lane);
        Sz = gemm_acc<256>(Sz, Az, 256, wih + (256 + n0) * 512 + 256, 512, lane);
        Sz = gemm_acc<256>(Sz, Ah, 256, whh + (256 + n0) * 256, 256, lane);
        v8f Gin = {0, 0, 0, 0, 0, 0, 0, 0};         // gi_n (u only)
        Gin = gemm_acc<256>(Gin, Ax, 256, wih + (512 + n0) * 512, 512, lane);
        Gin = gemm_acc<256>(Gin, Az, 256, wih + (512 + n0) * 512 + 256, 512, lane);
        v8f Ghn = {0, 0, 0, 0, 0, 0, 0, 0};         // gh_n separate (for r * gh_n)
        Ghn = gemm_acc<256>(Ghn, Ah, 256, whh + (512 + n0) * 256, 256, lane);

        int n = n0 + cn;
#pragma unroll
        for (int r = 0; r < 8; ++r) {
          float rg = sigmoidf_(Sr[r]);
          float zg = sigmoidf_(Sz[r]);
          float ng = tanhf(Gin[r] + rg * Ghn[r]);
          float hn = (1.0f - zg) * ng + zg * hreg[ml][i][r];
          hreg[ml][i][r] = hn;
          hnxt[(mt * 16 + mb + r) * 256 + n] = (_Float16)hn;
        }
      }
    }
    __syncthreads();  // hnxt complete before next step reads it as hcur
  }  // t loop

  // out = concat([phi_z_last, h_fin], -1): [B, 2H] f32
  for (int e = tid; e < BM * 256; e += 512) {
    int m = e >> 8, c = e & 255;
    out[(size_t)(r0 + m) * 512 + c] = (float)phiz[e];
  }
#pragma unroll
  for (int ml = 0; ml < 2; ++ml)
#pragma unroll
    for (int i = 0; i < 2; ++i) {
      int n = cg * 32 + i * 16 + cn;
      int m0 = (mh * 2 + ml) * 16 + mb;
#pragma unroll
      for (int r = 0; r < 8; ++r)
        out[(size_t)(r0 + m0 + r) * 512 + 256 + n] = hreg[ml][i][r];
    }
}

// ---------------------------------------------------------------------------
extern "C" void kernel_launch(void* const* d_in, const int* in_sizes, int n_in,
                              void* d_out, int out_size, void* d_ws, size_t ws_size,
                              hipStream_t stream) {
  const float* x   = (const float*)d_in[0];
  const float* h0  = (const float*)d_in[1];
  const float* eps = (const float*)d_in[2];
  const float* Wx1 = (const float*)d_in[3];
  const float* bx1 = (const float*)d_in[4];
  const float* Wx2 = (const float*)d_in[5];
  const float* bx2 = (const float*)d_in[6];
  const float* Wz  = (const float*)d_in[7];
  const float* bz  = (const float*)d_in[8];
  const float* We1 = (const float*)d_in[9];
  const float* be1 = (const float*)d_in[10];
  const float* We2 = (const float*)d_in[11];
  const float* be2 = (const float*)d_in[12];
  const float* Wem = (const float*)d_in[13];
  const float* bem = (const float*)d_in[14];
  const float* Wes = (const float*)d_in[15];
  const float* bes = (const float*)d_in[16];
  // prior branch (d_in[17..22]) is dead in the reference output
  const float* Wih = (const float*)d_in[23];
  const float* Whh = (const float*)d_in[24];

  // f16 weight arena in workspace (~1.75 MB)
  _Float16* w   = (_Float16*)d_ws;
  _Float16* wx1 = w;
  _Float16* wx2 = wx1 + 256 * 64;
  _Float16* wz  = wx2 + 256 * 256;
  _Float16* we1 = wz + 256 * 64;
  _Float16* we2 = we1 + 256 * 512;
  _Float16* wem = we2 + 256 * 256;
  _Float16* wes = wem + 64 * 256;
  _Float16* wih = wes + 64 * 256;
  _Float16* whh = wih + 768 * 512;

  struct Cv { const float* s; _Float16* d; int n; };
  Cv cv[9] = {
      {Wx1, wx1, 256 * 64},  {Wx2, wx2, 256 * 256}, {Wz, wz, 256 * 64},
      {We1, we1, 256 * 512}, {We2, we2, 256 * 256}, {Wem, wem, 64 * 256},
      {Wes, wes, 64 * 256},  {Wih, wih, 768 * 512}, {Whh, whh, 768 * 256}};
  for (int i = 0; i < 9; ++i) {
    int nblk = (cv[i].n / 4 + 255) / 256;
    f32_to_f16_kernel<<<nblk, 256, 0, stream>>>(cv[i].s, cv[i].d, cv[i].n);
  }

  // Allow 192KB dynamic LDS per WG (CDNA5 WGP: up to 320KB per workgroup).
  // Not a stream op; deterministic on every call. Ignore errors.
  (void)hipFuncSetAttribute((const void*)vrnn_kernel,
                            hipFuncAttributeMaxDynamicSharedMemorySize, 196608);

  vrnn_kernel<<<Bb / BM, 512, 196608, stream>>>(
      x, h0, eps, bx1, bx2, bz, be1, be2, bem, bes,
      wx1, wx2, wz, we1, we2, wem, wes, wih, whh, (float*)d_out);
}
